// TranformerEncoder_44641890075145
// MI455X (gfx1250) — matrix operand order repaired and checked by reading, not compiled, not run on hardware
//
#include <hip/hip_runtime.h>
#include <hip/hip_bf16.h>
#include <math.h>

typedef __attribute__((ext_vector_type(16))) __bf16 v16bf;
typedef __attribute__((ext_vector_type(8)))  __bf16 v8bf;
typedef __attribute__((ext_vector_type(4)))  __bf16 v4bf;
typedef __attribute__((ext_vector_type(8)))  float  v8f;
typedef __attribute__((ext_vector_type(4)))  unsigned int v4u;
typedef __attribute__((ext_vector_type(8)))  int    v8i;
typedef __attribute__((ext_vector_type(4)))  int    v4i;

#define DEV __device__ __forceinline__

#if defined(__has_builtin)
#if __has_builtin(__builtin_amdgcn_tensor_load_to_lds) && __has_builtin(__builtin_amdgcn_s_wait_tensorcnt)
#define HAS_TDM 1
#else
#define HAS_TDM 0
#endif
#else
#define HAS_TDM 0
#endif

DEV v16bf join8(v8bf a, v8bf b) {
  return __builtin_shufflevector(a, b, 0,1,2,3,4,5,6,7,8,9,10,11,12,13,14,15);
}
DEV v16bf ld16(const __bf16* p) {
  return join8(*(const v8bf*)p, *(const v8bf*)(p + 8));
}
DEV v8f wmma_bf16(v16bf a, v16bf b, v8f c) {
  // D = A(16x32 bf16) * B(32x16 bf16) + C(16x16 f32)
  return __builtin_amdgcn_wmma_f32_16x16x32_bf16(false, a, false, b, (short)0, c, false, false);
}

#if HAS_TDM
// Issue a 2D TDM tile load (global -> LDS), D# per CDNA5 ISA §8.3/8.4.
// tile_d0 elems contiguous per row, tile_d1 rows, row stride stride0 (elems),
// 2-byte elements. Caller gates to a single wave; sync via s_wait_tensorcnt.
DEV void tdm_load_2d(unsigned lds_off, const void* gaddr,
                     unsigned tile_d0, unsigned tile_d1,
                     unsigned tensor_d0, unsigned tensor_d1,
                     unsigned long long stride0) {
  unsigned long long ga = (unsigned long long)gaddr;
  v4u g0;
  g0[0] = 1u;                                   // count=1, user mode
  g0[1] = lds_off;                              // D#.lds_addr
  g0[2] = (unsigned)ga;                         // D#.global_addr[31:0]
  g0[3] = ((unsigned)(ga >> 32) & 0x01FFFFFFu)  // D#.global_addr[56:32]
          | 0x80000000u;                        // type=2 ("image")
  v8i g1;
  g1[0] = 0x00010000;                           // data_size=1 (2 bytes), mask=0
  g1[1] = (int)((tensor_d0 & 0xFFFFu) << 16);   // tensor_dim0[15:0]
  g1[2] = (int)((tensor_d0 >> 16) | ((tensor_d1 & 0xFFFFu) << 16));
  g1[3] = (int)((tensor_d1 >> 16) | (tile_d0 << 16));   // tile_dim0
  g1[4] = (int)(tile_d1 & 0xFFFFu);             // tile_dim1 (tile_dim2=0)
  g1[5] = (int)(unsigned)(stride0 & 0xFFFFFFFFull);     // dim0_stride[31:0]
  g1[6] = (int)(unsigned)((stride0 >> 32) & 0xFFFFull); // dim0_stride[47:32]
  g1[7] = 0;
  v4i z4 = {};
  v8i z8 = {};
  __builtin_amdgcn_tensor_load_to_lds(g0, g1, z4, z4, z8, 0);
}
#endif

// Scalar (SGPR-compare) test "is this wave 0", so TDM issue is a true
// uniform branch (s_cbranch), not an EXEC-masked region (TDM ignores EXEC).
DEV bool is_wave0() {
  return __builtin_amdgcn_readfirstlane((int)(threadIdx.x >> 5)) == 0;
}

// ---------------------------------------------------------------- convert
__global__ __launch_bounds__(256)
void cvt_kernel(const float* __restrict__ in, __bf16* __restrict__ out, int n) {
  int i = (blockIdx.x * blockDim.x + threadIdx.x) * 4;
  if (i >= n) return;
  float4 f = *(const float4*)(in + i);
  v4bf o;
  o[0] = (__bf16)f.x; o[1] = (__bf16)f.y; o[2] = (__bf16)f.z; o[3] = (__bf16)f.w;
  *(v4bf*)(out + i) = o;
}

// ---------------------------------------------------------------- GEMM
enum { EPI_F32 = 0, EPI_GELU = 1, EPI_QKV = 2 };

// C[M,Nt] = A[M,K](bf16) @ W[K,Nt](bf16) + bias, block tile 128x64, BK=32.
// Double-buffered LDS, one barrier per K-step: TDM for tile kk+1 overlaps
// the WMMA compute of tile kk. 8 waves, each owns a 32x32 subtile (2x2 WMMA).
template<int EPI>
__global__ __launch_bounds__(256)
void gemm_bf16_kernel(const __bf16* __restrict__ A, const __bf16* __restrict__ W,
                      const float* __restrict__ bias, int M, int Nt, int K,
                      float* __restrict__ Cf, __bf16* __restrict__ Cb,
                      __bf16* __restrict__ Qo, __bf16* __restrict__ Ko,
                      __bf16* __restrict__ Vo, int Nseq) {
  __shared__ __bf16 lds_a[2][128 * 32];   // A tiles, row-major [128][32]
  __shared__ __bf16 lds_bt[2][64 * 32];   // W tiles transposed [n][k]

  const int tid = threadIdx.x;
  const int w = tid >> 5, L = tid & 31, lg = L >> 4, ln = L & 15;
  const int wm = w & 3, wn = w >> 2;
  const int bm = blockIdx.y * 128;
  const int bn = blockIdx.x * 64;

  v8f acc[2][2] = {};

  const int nk = K >> 5;
  const int bkr = tid >> 3, bnc = (tid & 7) * 8;   // W: 8 threads/row, 8 bf16 each
  const int arow = tid >> 1, ahalf = tid & 1;      // A (fallback): 2 threads/row
  (void)arow; (void)ahalf;

  auto stage = [&](int kk, int buf) {
#if HAS_TDM
    if (is_wave0()) {
      // A tile 128 rows x 32 cols, row stride K, via Tensor Data Mover
      tdm_load_2d((unsigned)(unsigned long long)(void*)lds_a[buf],
                  A + (size_t)bm * K + kk * 32,
                  /*tile_d0=*/32, /*tile_d1=*/128,
                  /*tensor_d0=*/(unsigned)K, /*tensor_d1=*/(unsigned)M,
                  /*stride0=*/(unsigned long long)K);
    }
#else
    const __bf16* src = A + (size_t)(bm + arow) * K + kk * 32 + ahalf * 16;
    *(v8bf*)(lds_a[buf] + arow * 32 + ahalf * 16)     = *(const v8bf*)src;
    *(v8bf*)(lds_a[buf] + arow * 32 + ahalf * 16 + 8) = *(const v8bf*)(src + 8);
#endif
    v8bf b0 = *(const v8bf*)(W + (size_t)(kk * 32 + bkr) * Nt + bn + bnc);
#pragma unroll
    for (int i = 0; i < 8; ++i) lds_bt[buf][(bnc + i) * 32 + bkr] = b0[i];
  };

  stage(0, 0);
  for (int kk = 0; kk < nk; ++kk) {
    const int cur = kk & 1;
#if HAS_TDM
    __builtin_amdgcn_s_wait_tensorcnt(0);
#endif
    __syncthreads();
    if (kk + 1 < nk) stage(kk + 1, cur ^ 1);   // overlaps compute below

    v16bf af[2], bw[2];
#pragma unroll
    for (int mt = 0; mt < 2; ++mt) {
      // 16-bit A 16x32 layout: lane holds row ln, K = lg*16 + 0..15 (contiguous)
      const __bf16* p = lds_a[cur] + (wm * 32 + mt * 16 + ln) * 32;
      af[mt] = ld16(p + lg * 16);
    }
#pragma unroll
    for (int nt = 0; nt < 2; ++nt)
      // B 32x16 layout: lane holds col ln, K = lg*16 + 0..15 (contiguous)
      bw[nt] = ld16(lds_bt[cur] + (wn * 32 + nt * 16 + ln) * 32 + lg * 16);
#pragma unroll
    for (int mt = 0; mt < 2; ++mt)
#pragma unroll
      for (int nt = 0; nt < 2; ++nt)
        acc[mt][nt] = wmma_bf16(af[mt], bw[nt], acc[mt][nt]);
  }

  // epilogue: C element (v, lane) -> row v + 8*lg, col ln
#pragma unroll
  for (int mt = 0; mt < 2; ++mt) {
#pragma unroll
    for (int nt = 0; nt < 2; ++nt) {
      const int gn = bn + wn * 32 + nt * 16 + ln;
      const float bv = bias[gn];
#pragma unroll
      for (int v = 0; v < 8; ++v) {
        const int gm = bm + wm * 32 + mt * 16 + v + 8 * lg;
        float val = acc[mt][nt][v] + bv;
        if (EPI == EPI_F32) {
          Cf[(size_t)gm * Nt + gn] = val;
        } else if (EPI == EPI_GELU) {
          float gl = 0.5f * val * (1.0f + erff(val * 0.70710678118f));
          Cb[(size_t)gm * Nt + gn] = (__bf16)gl;
        } else { // EPI_QKV: scatter to per-head q/k/v [B,H,N,64]
          int s = gn >> 9, hh = (gn >> 6) & 7, e = gn & 63;
          __bf16* dst = (s == 0) ? Qo : (s == 1) ? Ko : Vo;
          int bb = gm / Nseq, nn = gm % Nseq;
          dst[(((size_t)bb * 8 + hh) * Nseq + nn) * 64 + e] = (__bf16)val;
        }
      }
    }
  }
}

// ---------------------------------------------------------------- flash attention
// block = 128 q rows of one (b,h); each of 8 waves owns 16 q rows (16x64 output),
// streams K/V in 32-row tiles with online softmax. Double-buffered K/V tiles,
// one barrier per tile; TDM K-load for tile kt+1 overlaps compute of tile kt.
// Post-softmax *0.125 scaling folded into the final normalize.
__global__ __launch_bounds__(256)
void flash_attn_kernel(const __bf16* __restrict__ Q, const __bf16* __restrict__ K,
                       const __bf16* __restrict__ V, __bf16* __restrict__ attb,
                       int Nseq) {
  __shared__ __bf16 lds_k[2][32 * 64];   // K tiles row-major [krow][hd]
  __shared__ __bf16 lds_vt[2][64 * 32];  // V tiles transposed [hd][krow]
  __shared__ __bf16 lds_p[8][16 * 32];   // per-wave P scratch (C-layout -> A-layout)

  const int bh = blockIdx.x;
  const int b = bh >> 3, h = bh & 7;
  const int tid = threadIdx.x, w = tid >> 5, L = tid & 31, lg = L >> 4, ln = L & 15;
  const int q0 = blockIdx.y * 128 + w * 16;

  // preload this wave's 16x64 q tile as two A-fragments (hd halves)
  const __bf16* qp = Q + ((size_t)bh * Nseq + q0 + ln) * 64;
  v16bf qa[2];
#pragma unroll
  for (int ks = 0; ks < 2; ++ks)
    qa[ks] = ld16(qp + ks * 32 + lg * 16);

  // all-ones B fragment: rowsum(P) = P @ ones via the matrix pipe
  v16bf ones;
#pragma unroll
  for (int i = 0; i < 16; ++i) ones[i] = (__bf16)1.0f;

  const __bf16* Kp = K + (size_t)bh * Nseq * 64;
  const __bf16* Vp = V + (size_t)bh * Nseq * 64;

  v8f o[4] = {};
  float mrun[8], lrun[8];
#pragma unroll
  for (int v = 0; v < 8; ++v) { mrun[v] = -3.0e38f; lrun[v] = 0.0f; }

  const int skr = tid >> 3, snc = (tid & 7) * 8;
  const int nkt = Nseq >> 5;

  auto stage = [&](int kt, int buf) {
#if HAS_TDM
    if (is_wave0()) {
      // K tile 32 rows x 64 cols, contiguous rows, via Tensor Data Mover
      tdm_load_2d((unsigned)(unsigned long long)(void*)lds_k[buf],
                  Kp + (size_t)kt * 32 * 64,
                  /*tile_d0=*/64, /*tile_d1=*/32,
                  /*tensor_d0=*/64, /*tensor_d1=*/(unsigned)Nseq,
                  /*stride0=*/64ull);
    }
#else
    *(v8bf*)(lds_k[buf] + skr * 64 + snc) =
        *(const v8bf*)(Kp + (size_t)(kt * 32 + skr) * 64 + snc);
#endif
    v8bf vv = *(const v8bf*)(Vp + (size_t)(kt * 32 + skr) * 64 + snc);
#pragma unroll
    for (int i = 0; i < 8; ++i) lds_vt[buf][(snc + i) * 32 + skr] = vv[i];
  };

  stage(0, 0);
  for (int kt = 0; kt < nkt; ++kt) {
    const int cur = kt & 1;
#if HAS_TDM
    __builtin_amdgcn_s_wait_tensorcnt(0);
#endif
    __syncthreads();
    if (kt + 1 < nkt) stage(kt + 1, cur ^ 1);   // overlaps compute below

    // S(16x32) = q(16x64) @ k_tile^T : B-frag k-dim runs along hd (contiguous)
    v8f s[2];
#pragma unroll
    for (int ct = 0; ct < 2; ++ct) {
      v8f a_ = {};
#pragma unroll
      for (int ks = 0; ks < 2; ++ks) {
        v16bf bfr = ld16(lds_k[cur] + (ct * 16 + ln) * 64 + ks * 32 + lg * 16);
        a_ = wmma_bf16(qa[ks], bfr, a_);
      }
      s[ct] = a_;
    }

    // online softmax max; row of element v is (v + 8*lg), reduced over 16 lanes
    float sc[8];
#pragma unroll
    for (int v = 0; v < 8; ++v) {
      float mx = fmaxf(s[0][v], s[1][v]);
#pragma unroll
      for (int msk = 8; msk >= 1; msk >>= 1) mx = fmaxf(mx, __shfl_xor(mx, msk, 32));
      float mnew  = fmaxf(mrun[v], mx);
      sc[v] = __expf(mrun[v] - mnew);
      mrun[v] = mnew;
      s[0][v] = __expf(s[0][v] - mnew);
      s[1][v] = __expf(s[1][v] - mnew);
    }
#pragma unroll
    for (int nc = 0; nc < 4; ++nc)
#pragma unroll
      for (int v = 0; v < 8; ++v) o[nc][v] *= sc[v];

    // P: C-layout -> LDS row-major -> A-layout fragment (per-wave region)
#pragma unroll
    for (int ct = 0; ct < 2; ++ct)
#pragma unroll
      for (int v = 0; v < 8; ++v)
        lds_p[w][(v + 8 * lg) * 32 + ct * 16 + ln] = (__bf16)s[ct][v];

    const __bf16* pr = &lds_p[w][ln * 32];
    v16bf pa = ld16(pr + lg * 16);

    // rowsum(P) on the matrix pipe: every column of the result equals the row sum
    v8f zz = {};
    v8f rsv = wmma_bf16(pa, ones, zz);
#pragma unroll
    for (int v = 0; v < 8; ++v) lrun[v] = lrun[v] * sc[v] + rsv[v];

#pragma unroll
    for (int nc = 0; nc < 4; ++nc) {
      v16bf bv = ld16(lds_vt[cur] + (nc * 16 + ln) * 32 + lg * 16);
      o[nc] = wmma_bf16(pa, bv, o[nc]);
    }
  }

  // normalize (softmax denom) * 0.125 scaling, write att [B,N,D] bf16
  __bf16* op = attb + ((size_t)b * Nseq + q0) * 512 + h * 64;
  float inv[8];
#pragma unroll
  for (int v = 0; v < 8; ++v) inv[v] = 0.125f / lrun[v];
#pragma unroll
  for (int nc = 0; nc < 4; ++nc)
#pragma unroll
    for (int v = 0; v < 8; ++v)
      op[(size_t)(v + 8 * lg) * 512 + nc * 16 + ln] = (__bf16)(o[nc][v] * inv[v]);
}

// ---------------------------------------------------------------- residual + LN
template<bool WB>
__global__ __launch_bounds__(256)
void ln_residual_kernel(const float* __restrict__ xr, const float* __restrict__ dl,
                        const float* __restrict__ g, const float* __restrict__ be,
                        float* __restrict__ outf, __bf16* __restrict__ outb) {
  const int row = blockIdx.x;
  const int t = threadIdx.x;                 // D=512, 2 elements/thread
  const size_t base = (size_t)row * 512 + t * 2;
  float s0 = xr[base] + dl[base];
  float s1 = xr[base + 1] + dl[base + 1];
  float sum = s0 + s1, sq = s0 * s0 + s1 * s1;
#pragma unroll
  for (int m = 16; m >= 1; m >>= 1) {
    sum += __shfl_xor(sum, m, 32);
    sq  += __shfl_xor(sq,  m, 32);
  }
  __shared__ float ssum[8], ssq[8];
  if ((t & 31) == 0) { ssum[t >> 5] = sum; ssq[t >> 5] = sq; }
  __syncthreads();
  sum = ssum[0]; sq = ssq[0];
#pragma unroll
  for (int i = 1; i < 8; ++i) { sum += ssum[i]; sq += ssq[i]; }
  const float mu  = sum * (1.0f / 512.0f);
  const float var = sq * (1.0f / 512.0f) - mu * mu;
  const float r = rsqrtf(var + 1e-5f);
  float y0 = (s0 - mu) * r * g[t * 2]     + be[t * 2];
  float y1 = (s1 - mu) * r * g[t * 2 + 1] + be[t * 2 + 1];
  outf[base] = y0; outf[base + 1] = y1;
  if (WB) { outb[base] = (__bf16)y0; outb[base + 1] = (__bf16)y1; }
}

// ---------------------------------------------------------------- launch
extern "C" void kernel_launch(void* const* d_in, const int* in_sizes, int n_in,
                              void* d_out, int out_size, void* d_ws, size_t ws_size,
                              hipStream_t stream) {
  (void)in_sizes; (void)n_in; (void)out_size; (void)ws_size;
  const float* x      = (const float*)d_in[0];
  const float* w_qkv  = (const float*)d_in[1];
  const float* b_qkv  = (const float*)d_in[2];
  const float* w_proj = (const float*)d_in[3];
  const float* b_proj = (const float*)d_in[4];
  const float* ln1_g  = (const float*)d_in[5];
  const float* ln1_b  = (const float*)d_in[6];
  const float* w1     = (const float*)d_in[7];
  const float* b1     = (const float*)d_in[8];
  const float* w2     = (const float*)d_in[9];
  const float* b2     = (const float*)d_in[10];
  const float* ln2_g  = (const float*)d_in[11];
  const float* ln2_b  = (const float*)d_in[12];

  const int B = 4, Nseq = 2048, D = 512, Hf = 2048;
  const int M = B * Nseq;

  char* ws = (char*)d_ws;
  size_t off = 0;
  auto alloc = [&](size_t bytes) -> char* {
    char* p = ws + off;
    off = (off + bytes + 255) & ~(size_t)255;
    return p;
  };
  __bf16* xb     = (__bf16*)alloc((size_t)M * D * 2);
  __bf16* wqkvb  = (__bf16*)alloc((size_t)D * 3 * D * 2);
  __bf16* wprojb = (__bf16*)alloc((size_t)D * D * 2);
  __bf16* w1b    = (__bf16*)alloc((size_t)D * Hf * 2);
  __bf16* w2b    = (__bf16*)alloc((size_t)Hf * D * 2);
  __bf16* qb     = (__bf16*)alloc((size_t)M * D * 2);
  __bf16* kb     = (__bf16*)alloc((size_t)M * D * 2);
  __bf16* vb     = (__bf16*)alloc((size_t)M * D * 2);
  __bf16* attb   = (__bf16*)alloc((size_t)M * D * 2);
  float*  projf  = (float*) alloc((size_t)M * D * 4);
  float*  x1f    = (float*) alloc((size_t)M * D * 4);
  __bf16* x1b    = (__bf16*)alloc((size_t)M * D * 2);
  __bf16* hb     = (__bf16*)alloc((size_t)M * Hf * 2);
  float*  f2f    = (float*) alloc((size_t)M * D * 4);

  auto cvt = [&](const float* src, __bf16* dst, int n) {
    cvt_kernel<<<dim3((n / 4 + 255) / 256), dim3(256), 0, stream>>>(src, dst, n);
  };
  cvt(x, xb, M * D);
  cvt(w_qkv, wqkvb, D * 3 * D);
  cvt(w_proj, wprojb, D * D);
  cvt(w1, w1b, D * Hf);
  cvt(w2, w2b, Hf * D);

  // qkv = x @ w_qkv + b_qkv, scatter to per-head q/k/v
  gemm_bf16_kernel<EPI_QKV><<<dim3(3 * D / 64, M / 128), 256, 0, stream>>>(
      xb, wqkvb, b_qkv, M, 3 * D, D, nullptr, nullptr, qb, kb, vb, Nseq);

  // att = softmax(q k^T) * 0.125 @ v  (fused, flash-style)
  flash_attn_kernel<<<dim3(B * 8, Nseq / 128), 256, 0, stream>>>(qb, kb, vb, attb, Nseq);

  // proj
  gemm_bf16_kernel<EPI_F32><<<dim3(D / 64, M / 128), 256, 0, stream>>>(
      attb, wprojb, b_proj, M, D, D, projf, nullptr, nullptr, nullptr, nullptr, Nseq);

  // x1 = LN(x + proj)
  ln_residual_kernel<true><<<dim3(M), 256, 0, stream>>>(x, projf, ln1_g, ln1_b, x1f, x1b);

  // h = gelu(x1 @ w1 + b1)
  gemm_bf16_kernel<EPI_GELU><<<dim3(Hf / 64, M / 128), 256, 0, stream>>>(
      x1b, w1b, b1, M, Hf, D, nullptr, hb, nullptr, nullptr, nullptr, Nseq);

  // f2 = h @ w2 + b2
  gemm_bf16_kernel<EPI_F32><<<dim3(D / 64, M / 128), 256, 0, stream>>>(
      hb, w2b, b2, M, D, Hf, f2f, nullptr, nullptr, nullptr, nullptr, Nseq);

  // out = LN(x1 + f2)
  ln_residual_kernel<false><<<dim3(M), 256, 0, stream>>>(x1f, f2f, ln2_g, ln2_b,
                                                         (float*)d_out, nullptr);
}
